// HeteroForecastGATConv_85822036509293
// MI455X (gfx1250) — matrix operand order repaired and checked by reading, not compiled, not run on hardware
//
#include <hip/hip_runtime.h>
#include <hip/hip_bf16.h>

typedef __attribute__((ext_vector_type(16))) _Float16 v16h;
typedef __attribute__((ext_vector_type(8)))  float    v8f;

#define NT_   100000
#define NC_   25000
#define ETT_  800000
#define ECT_  400000
#define HEADS 4
#define CH    32
#define HIDV  128
#define SLOPE_ 0.2f

// ---------------------------------------------------------------------------
// Pre-pack a weight matrix W[K,128] (f32, row-major) into f16 WMMA-B fragment
// order: P[idx], idx = ((k0/32)*8 + colTile)*512 + lane*16 + e, where the
// fragment element mapping is  k = k0 + 16*(lane>>4) + e,  n = colTile*16 +
// (lane&15).  One 32B vector load per lane then yields a whole B fragment.
// ---------------------------------------------------------------------------
__global__ __launch_bounds__(256) void pack_w_f16(
    const float* __restrict__ W, _Float16* __restrict__ P, int K)
{
    int total = K * HIDV;
    for (int idx = blockIdx.x * blockDim.x + threadIdx.x; idx < total;
         idx += gridDim.x * blockDim.x) {
        int e    = idx & 15;
        int lane = (idx >> 4) & 31;
        int tile = idx >> 9;
        int k = (tile >> 3) * 32 + 16 * (lane >> 4) + e;
        int n = (tile & 7) * 16 + (lane & 15);
        P[idx] = (_Float16)W[k * HIDV + n];
    }
}

// ---------------------------------------------------------------------------
// WMMA GEMM: Out[M,128] = act(X[M,K] @ W[K,128] + bias), W pre-packed f16.
// 256 threads = 8 wave32; wave w owns rows [16*bx,+16) x cols [16*w,+16).
// Per K-step: 4x b128 (A, cvt to f16) + 2x b128 (packed B) + 1 wmma.
// ---------------------------------------------------------------------------
template <int K>
__global__ __launch_bounds__(256) void gemm_wmma_f16(
    const float* __restrict__ X, const _Float16* __restrict__ Wp,
    const float* __restrict__ bias, float* __restrict__ Out,
    int M, int act)
{
    const int N    = HIDV;
    const int lane = threadIdx.x & 31;
    const int wave = threadIdx.x >> 5;
    const int row0 = blockIdx.x << 4;
    const int col0 = wave << 4;
    const int g    = lane >> 4;      // lane group (0/1)
    const int mr   = lane & 15;
    int row = row0 + mr; if (row >= M) row = M - 1;   // clamp (tail tiles)
    const int n = col0 + mr;

    v8f acc = {};
    #pragma unroll
    for (int k0 = 0; k0 < K; k0 += 32) {
        v16h a;
        // A fragment 16x32 f16: lane (g,mr): element e -> k = 16*g+e
        const float* ap = X + (size_t)row * K + k0 + 16 * g;
        #pragma unroll
        for (int h2 = 0; h2 < 2; ++h2) {
            const float4* p4 = (const float4*)(ap + 8 * h2);
            float4 v0 = p4[0];
            float4 v1 = p4[1];
            a[8*h2+0] = (_Float16)v0.x; a[8*h2+1] = (_Float16)v0.y;
            a[8*h2+2] = (_Float16)v0.z; a[8*h2+3] = (_Float16)v0.w;
            a[8*h2+4] = (_Float16)v1.x; a[8*h2+5] = (_Float16)v1.y;
            a[8*h2+6] = (_Float16)v1.z; a[8*h2+7] = (_Float16)v1.w;
        }
        // B fragment: one 32B vector load from the packed-fragment layout.
        v16h b = *((const v16h*)(Wp + (size_t)((((k0 >> 5) << 3) + wave) << 9)) + lane);

        acc = __builtin_amdgcn_wmma_f32_16x16x32_f16(
            false, a, false, b, (short)0, acc, false, false);
    }

    const float bv = bias ? bias[n] : 0.0f;
    #pragma unroll
    for (int r = 0; r < 8; ++r) {
        int orow = row0 + r + 8 * g;          // C/D layout: VGPR r -> M = r + 8*g
        if (orow < M) {
            float v = acc[r] + bv;
            if (act) v = v > 0.0f ? v : 0.0f;
            Out[(size_t)orow * N + n] = v;
        }
    }
}

// a[n,h] = sum_c hm[n, h*32+c] * att[h,c]
__global__ __launch_bounds__(256) void att_logits(
    const float* __restrict__ hm, const float* __restrict__ att,
    float* __restrict__ out, int N)
{
    int t = blockIdx.x * blockDim.x + threadIdx.x;
    if (t >= N * HEADS) return;
    int n = t >> 2, h = t & 3;
    const float* hp = hm + (size_t)n * HIDV + h * CH;
    const float* ap = att + h * CH;
    float s = 0.f;
    #pragma unroll
    for (int c = 0; c < CH; ++c) s += hp[c] * ap[c];
    out[t] = s;
}

// float atomic max via ordered-int trick; init bits 0xFFFFFFFF act as -inf.
__device__ inline void atomicMaxF(float* addr, float val) {
    if (val >= 0.0f) atomicMax((int*)addr, __float_as_int(val));
    else             atomicMin((unsigned int*)addr, __float_as_uint(val));
}

// e = LeakyReLU(a_s[src]+a_d[dst]); segment max into mbuf. Edges i>=E are
// implicit self-loops (src=dst=i-E).
__global__ __launch_bounds__(256) void edge_logits_max(
    const float* __restrict__ as_, const float* __restrict__ ad_,
    const int* __restrict__ src, const int* __restrict__ dst,
    int E, int nloops, float* __restrict__ ebuf, float* __restrict__ mbuf)
{
    int t = blockIdx.x * blockDim.x + threadIdx.x;
    if (t >= (E + nloops) * HEADS) return;
    int i = t >> 2, h = t & 3;
    int s, d;
    if (i < E) { s = src[i]; d = dst[i]; } else { s = i - E; d = s; }
    float e = as_[s * HEADS + h] + ad_[d * HEADS + h];
    e = e > 0.f ? e : SLOPE_ * e;
    ebuf[t] = e;
    atomicMaxF(&mbuf[d * HEADS + h], e);
}

// ex = exp(e - m[dst]); segment sum into den; ebuf overwritten with ex.
__global__ __launch_bounds__(256) void edge_exp_sum(
    float* __restrict__ ebuf, const int* __restrict__ dst,
    int E, int nloops, const float* __restrict__ mbuf, float* __restrict__ den)
{
    int t = blockIdx.x * blockDim.x + threadIdx.x;
    if (t >= (E + nloops) * HEADS) return;
    int i = t >> 2, h = t & 3;
    int d = (i < E) ? dst[i] : (i - E);
    float ex = __expf(ebuf[t] - mbuf[d * HEADS + h]);
    ebuf[t] = ex;
    atomicAdd(&den[d * HEADS + h], ex);
}

// agg[dst, j] += h_src[src, j] * (ex / den[dst, head(j)])
__global__ __launch_bounds__(256) void edge_message(
    const float* __restrict__ hsrc, const float* __restrict__ ebuf,
    const float* __restrict__ den, const int* __restrict__ src,
    const int* __restrict__ dst, int E, int nloops, float* __restrict__ agg)
{
    long long t = (long long)blockIdx.x * blockDim.x + threadIdx.x;
    if (t >= (long long)(E + nloops) * HIDV) return;
    int i = (int)(t >> 7);
    int j = (int)(t & 127);
    int h = j >> 5;
    int s, d;
    if (i < E) { s = src[i]; d = dst[i]; } else { s = i - E; d = s; }
    float alpha = ebuf[i * HEADS + h] / den[d * HEADS + h];
    atomicAdd(&agg[(size_t)d * HIDV + j], hsrc[(size_t)s * HIDV + j] * alpha);
}

// x = relu(0.25*(agg_sd+b_sd) + 0.25*(agg_ds+b_ds) + 0.5*(agg_ct+b_ct) + res)
__global__ __launch_bounds__(256) void combine_skip_relu(
    const float* __restrict__ a_sd, const float* __restrict__ a_ds,
    const float* __restrict__ a_ct, const float* __restrict__ b_sd,
    const float* __restrict__ b_ds, const float* __restrict__ b_ct,
    const float* __restrict__ res, float* __restrict__ xo)
{
    int t = blockIdx.x * blockDim.x + threadIdx.x;
    if (t >= NT_ * HIDV) return;
    int j = t & 127;
    float v = 0.25f * (a_sd[t] + b_sd[j]) + 0.25f * (a_ds[t] + b_ds[j])
            + 0.5f  * (a_ct[t] + b_ct[j]) + res[t];
    xo[t] = v > 0.f ? v : 0.f;
}

extern "C" void kernel_launch(void* const* d_in, const int* in_sizes, int n_in,
                              void* d_out, int out_size, void* d_ws, size_t ws_size,
                              hipStream_t stream)
{
    (void)in_sizes; (void)n_in; (void)out_size; (void)ws_size;
    const float* x_t      = (const float*)d_in[0];
    const float* x_c      = (const float*)d_in[1];
    const int*   e_tt     = (const int*)d_in[2];   // [2, ETT] row-major
    const int*   ct_src   = (const int*)d_in[3];
    const int*   ct_dst   = (const int*)d_in[4];
    const float* W_pre_t  = (const float*)d_in[5];
    const float* b_pre_t  = (const float*)d_in[6];
    const float* W_pre_c  = (const float*)d_in[7];
    const float* b_pre_c  = (const float*)d_in[8];
    const float* W_sd     = (const float*)d_in[9];
    const float* att_s_sd = (const float*)d_in[10];
    const float* att_d_sd = (const float*)d_in[11];
    const float* b_sd     = (const float*)d_in[12];
    const float* W_ds     = (const float*)d_in[13];
    const float* att_s_ds = (const float*)d_in[14];
    const float* att_d_ds = (const float*)d_in[15];
    const float* b_ds     = (const float*)d_in[16];
    const float* W_ct_s   = (const float*)d_in[17];
    const float* W_ct_d   = (const float*)d_in[18];
    const float* att_s_ct = (const float*)d_in[19];
    const float* att_d_ct = (const float*)d_in[20];
    const float* b_ct     = (const float*)d_in[21];
    const float* W_out    = (const float*)d_in[22];
    const float* b_out    = (const float*)d_in[23];

    char* wsp = (char*)d_ws;
    auto alloc = [&](size_t bytes) -> void* {
        char* p = wsp;
        wsp += (bytes + 255) & ~(size_t)255;
        return (void*)p;
    };
    const size_t fNT = (size_t)NT_ * HIDV * sizeof(float);
    const size_t fNC = (size_t)NC_ * HIDV * sizeof(float);
    const size_t fNH = (size_t)NT_ * HEADS * sizeof(float);

    float* h_t    = (float*)alloc(fNT);
    float* h_c    = (float*)alloc(fNC);
    float* hs     = (float*)alloc(fNT);
    float* hd     = (float*)alloc(fNT);
    float* hcs    = (float*)alloc(fNC);
    float* htd    = (float*)alloc(fNT);
    float* agg_sd = (float*)alloc(fNT);
    float* agg_ds = (float*)alloc(fNT);
    float* agg_ct = (float*)alloc(fNT);
    float* asb    = (float*)alloc(fNH);
    float* adb    = (float*)alloc(fNH);
    float* mb     = (float*)alloc(fNH);
    float* db     = (float*)alloc(fNH);
    float* exb    = (float*)alloc((size_t)(ETT_ + NT_) * HEADS * sizeof(float));
    // packed f16 weights (tiny; L2-resident)
    _Float16* Pp_t  = (_Float16*)alloc((size_t)64  * HIDV * 2);
    _Float16* Pp_c  = (_Float16*)alloc((size_t)32  * HIDV * 2);
    _Float16* P_sd  = (_Float16*)alloc((size_t)128 * HIDV * 2);
    _Float16* P_ds  = (_Float16*)alloc((size_t)128 * HIDV * 2);
    _Float16* P_cts = (_Float16*)alloc((size_t)128 * HIDV * 2);
    _Float16* P_ctd = (_Float16*)alloc((size_t)128 * HIDV * 2);
    _Float16* P_out = (_Float16*)alloc((size_t)128 * HIDV * 2);
    float* xbuf = hs;   // hs dead after sd message pass; reuse for combine out

    const dim3 blk(256);
    const int gT = (NT_ + 15) / 16;
    const int gC = (NC_ + 15) / 16;

    // Pack all weights to f16 fragment order (one-time, tiny)
    pack_w_f16<<<32, blk, 0, stream>>>(W_pre_t, Pp_t, 64);
    pack_w_f16<<<16, blk, 0, stream>>>(W_pre_c, Pp_c, 32);
    pack_w_f16<<<64, blk, 0, stream>>>(W_sd,   P_sd,  128);
    pack_w_f16<<<64, blk, 0, stream>>>(W_ds,   P_ds,  128);
    pack_w_f16<<<64, blk, 0, stream>>>(W_ct_s, P_cts, 128);
    pack_w_f16<<<64, blk, 0, stream>>>(W_ct_d, P_ctd, 128);
    pack_w_f16<<<64, blk, 0, stream>>>(W_out,  P_out, 128);

    // Dense projections (WMMA f16->f32)
    gemm_wmma_f16<64> <<<gT, blk, 0, stream>>>(x_t, Pp_t, b_pre_t, h_t, NT_, 1);
    gemm_wmma_f16<32> <<<gC, blk, 0, stream>>>(x_c, Pp_c, b_pre_c, h_c, NC_, 1);
    gemm_wmma_f16<128><<<gT, blk, 0, stream>>>(h_t, P_sd,  nullptr, hs,  NT_, 0);
    gemm_wmma_f16<128><<<gT, blk, 0, stream>>>(h_t, P_ds,  nullptr, hd,  NT_, 0);
    gemm_wmma_f16<128><<<gC, blk, 0, stream>>>(h_c, P_cts, nullptr, hcs, NC_, 0);
    gemm_wmma_f16<128><<<gT, blk, 0, stream>>>(h_t, P_ctd, nullptr, htd, NT_, 0);

    auto run_gat = [&](const float* hsrc, const float* hdst,
                       const float* att_s, const float* att_d,
                       const int* src, const int* dst, int E, int nl,
                       int Nsrc, float* agg) {
        att_logits<<<(Nsrc * HEADS + 255) / 256, blk, 0, stream>>>(hsrc, att_s, asb, Nsrc);
        att_logits<<<(NT_  * HEADS + 255) / 256, blk, 0, stream>>>(hdst, att_d, adb, NT_);
        hipMemsetAsync(mb,  0xFF, fNH, stream);   // -inf-equivalent bits for atomicMaxF
        hipMemsetAsync(db,  0,    fNH, stream);
        hipMemsetAsync(agg, 0,    fNT, stream);
        const int tot = (E + nl) * HEADS;
        edge_logits_max<<<(tot + 255) / 256, blk, 0, stream>>>(asb, adb, src, dst, E, nl, exb, mb);
        edge_exp_sum  <<<(tot + 255) / 256, blk, 0, stream>>>(exb, dst, E, nl, mb, db);
        const long long mt = (long long)(E + nl) * HIDV;
        edge_message<<<(int)((mt + 255) / 256), blk, 0, stream>>>(hsrc, exb, db, src, dst, E, nl, agg);
    };

    // (target->target) forward, reversed (each with self-loops), then bipartite ct
    run_gat(hs,  hs,  att_s_sd, att_d_sd, e_tt,        e_tt + ETT_, ETT_, NT_, NT_, agg_sd);
    run_gat(hd,  hd,  att_s_ds, att_d_ds, e_tt + ETT_, e_tt,        ETT_, NT_, NT_, agg_ds);
    run_gat(hcs, htd, att_s_ct, att_d_ct, ct_src,      ct_dst,      ECT_, 0,   NC_, agg_ct);

    combine_skip_relu<<<(NT_ * HIDV + 255) / 256, blk, 0, stream>>>(
        agg_sd, agg_ds, agg_ct, b_sd, b_ds, b_ct, h_t, xbuf);

    gemm_wmma_f16<128><<<gT, blk, 0, stream>>>(xbuf, P_out, b_out, (float*)d_out, NT_, 0);
}
